// MixtureOfExperts_74036646248849
// MI455X (gfx1250) — matrix operand order repaired and checked by reading, not compiled, not run on hardware
//
#include <hip/hip_runtime.h>
#include <math.h>

// Problem constants (match the reference)
#define B_   4
#define S_   4096
#define D_   1024
#define E_   8
#define H_   2048
#define NT   (B_ * S_)      // 16384 tokens
#define MAXT NT             // worst-case tokens per expert

// GEMM tiling
#define BM  128
#define BN  128
#define BK  64
#define LDK 88              // LDS row stride (bf16 elems): 176B, 16B-aligned, 16-way bank-conflict-free
#define TILE_ELEMS (BM * LDK)   // per A or B buffer

typedef __bf16 v16bf __attribute__((ext_vector_type(16)));
typedef float  v8f   __attribute__((ext_vector_type(8)));
typedef int    v4i_  __attribute__((vector_size(16)));

union Frag { int4 q[2]; v16bf v; };

#if defined(__has_builtin)
#if __has_builtin(__builtin_amdgcn_global_load_async_to_lds_b128)
#define ASYNC_LDS 1
#endif
#endif
#ifndef ASYNC_LDS
#define ASYNC_LDS 0
#endif

// 16-byte global -> LDS copy. Async DMA path on gfx1250 (no VGPR staging,
// tracked with ASYNCcnt); falls back to a plain staged copy otherwise.
__device__ __forceinline__ void copy16_g2l(const void* g, void* l) {
#if ASYNC_LDS
  v4i_* gg = (v4i_*)g;   // drop const (generic AS)
  __builtin_amdgcn_global_load_async_to_lds_b128(
      (__attribute__((address_space(1))) v4i_*)gg,
      (__attribute__((address_space(3))) v4i_*)l, 0, 0);
#else
  *(int4*)l = *(const int4*)g;
#endif
}

// Wait so that at most the 8 just-issued (next-tile) async loads remain.
__device__ __forceinline__ void wait_async_le8() {
#if ASYNC_LDS
  asm volatile("s_wait_asynccnt 0x8" ::: "memory");
#endif
}
__device__ __forceinline__ void wait_async_0() {
#if ASYNC_LDS
  asm volatile("s_wait_asynccnt 0x0" ::: "memory");
#endif
}

__device__ __forceinline__ unsigned short f2bf(float f) {
  unsigned int u = __float_as_uint(f);
  u += 0x7FFFu + ((u >> 16) & 1u);      // round-to-nearest-even
  return (unsigned short)(u >> 16);
}

__device__ __forceinline__ float gelu_exact(float x) {
  return 0.5f * x * (1.0f + erff(x * 0.70710678118654752440f));
}

// ---------------- init / conversion kernels ----------------

__global__ void moe_zero_cnt(int* cnt) {
  if (threadIdx.x < E_) cnt[threadIdx.x] = 0;
}

__global__ void moe_zero_f32(float* p, long long n) {
  long long i = (long long)blockIdx.x * blockDim.x + threadIdx.x;
  long long st = (long long)gridDim.x * blockDim.x;
  for (; i < n; i += st) p[i] = 0.0f;
}

__global__ void moe_cvt_x(const float* __restrict__ x, unsigned short* __restrict__ xb, long long n) {
  long long i = (long long)blockIdx.x * blockDim.x + threadIdx.x;
  long long st = (long long)gridDim.x * blockDim.x;
  for (; i < n; i += st) xb[i] = f2bf(x[i]);
}

// w1t[(e*H + h)*D + d] = bf16(W1[(e*D + d)*H + h])
__global__ void moe_cvt_w1(const float* __restrict__ W1, unsigned short* __restrict__ w1t, long long n) {
  long long i = (long long)blockIdx.x * blockDim.x + threadIdx.x;
  long long st = (long long)gridDim.x * blockDim.x;
  for (; i < n; i += st) {
    long long d = i % D_;
    long long t = i / D_;
    long long h = t % H_;
    long long e = t / H_;
    w1t[i] = f2bf(W1[(e * D_ + d) * H_ + h]);
  }
}

// w2t[(e*D + d)*H + h] = bf16(W2[(e*H + h)*D + d])
__global__ void moe_cvt_w2(const float* __restrict__ W2, unsigned short* __restrict__ w2t, long long n) {
  long long i = (long long)blockIdx.x * blockDim.x + threadIdx.x;
  long long st = (long long)gridDim.x * blockDim.x;
  for (; i < n; i += st) {
    long long h = i % H_;
    long long t = i / H_;
    long long d = t % D_;
    long long e = t / D_;
    w2t[i] = f2bf(W2[(e * H_ + h) * D_ + d]);
  }
}

// ---------------- router: one wave32 per token ----------------

__global__ __launch_bounds__(256)
void moe_router(const float* __restrict__ x, const float* __restrict__ Wr,
                const float* __restrict__ br,
                int* __restrict__ cnt, int* __restrict__ tokenId, float* __restrict__ tokenW) {
  int wave = threadIdx.x >> 5;
  int lane = threadIdx.x & 31;
  int tok  = blockIdx.x * 8 + wave;
  if (tok >= NT) return;

  float acc[E_];
#pragma unroll
  for (int e = 0; e < E_; ++e) acc[e] = 0.0f;

  const float* xr = x + (size_t)tok * D_;
  for (int d = lane; d < D_; d += 32) {
    float xv = xr[d];
#pragma unroll
    for (int e = 0; e < E_; ++e) acc[e] += xv * Wr[d * E_ + e];
  }
#pragma unroll
  for (int off = 16; off > 0; off >>= 1) {
#pragma unroll
    for (int e = 0; e < E_; ++e) acc[e] += __shfl_xor(acc[e], off, 32);
  }

  if (lane == 0) {
    float m = -1e30f;
#pragma unroll
    for (int e = 0; e < E_; ++e) { acc[e] += br[e]; m = fmaxf(m, acc[e]); }
    float p[E_]; float s = 0.0f;
#pragma unroll
    for (int e = 0; e < E_; ++e) { p[e] = expf(acc[e] - m); s += p[e]; }
#pragma unroll
    for (int e = 0; e < E_; ++e) p[e] /= s;

    int i0 = 0;
#pragma unroll
    for (int e = 1; e < E_; ++e) if (p[e] > p[i0]) i0 = e;
    int i1 = (i0 == 0) ? 1 : 0;
#pragma unroll
    for (int e = 0; e < E_; ++e) if (e != i0 && p[e] > p[i1]) i1 = e;

    float w0 = p[i0], w1 = p[i1];
    float inv = 1.0f / (w0 + w1 + 1e-8f);
    w0 *= inv; w1 *= inv;

    int pos0 = atomicAdd(&cnt[i0], 1);
    tokenId[i0 * MAXT + pos0] = tok;
    tokenW [i0 * MAXT + pos0] = w0;
    int pos1 = atomicAdd(&cnt[i1], 1);
    tokenId[i1 * MAXT + pos1] = tok;
    tokenW [i1 * MAXT + pos1] = w1;
  }
}

__global__ void moe_scan(const int* __restrict__ cnt, int* __restrict__ offs) {
  if (threadIdx.x == 0) {
    int a = 0;
    for (int e = 0; e < E_; ++e) { offs[e] = a; a += cnt[e]; }
    offs[E_] = a;
  }
}

// ---------------- GEMM1: Hbuf = GELU(X_gathered @ W1T + b1), bf16 out ----------------

__global__ __launch_bounds__(256)
void moe_gemm1(const unsigned short* __restrict__ xb, const unsigned short* __restrict__ w1t,
               const float* __restrict__ b1,
               const int* __restrict__ cnt, const int* __restrict__ offs,
               const int* __restrict__ tokenId,
               unsigned short* __restrict__ hbuf) {
  const int e = blockIdx.z;
  const int cntE = cnt[e];
  const int tileM = blockIdx.x * BM;
  if (tileM >= cntE) return;
  const int n0 = blockIdx.y * BN;
  const int* list = tokenId + e * MAXT;
  const int base = offs[e];

  // Layout: [A0 | B0 | A1 | B1], each TILE_ELEMS bf16.
  __shared__ __align__(16) unsigned short smem[4 * TILE_ELEMS];

  const int wave = threadIdx.x >> 5;
  const int lane = threadIdx.x & 31;
  const int wm = wave >> 2;          // 0..1  (64-row slice)
  const int wn = wave & 3;           // 0..3  (32-col slice)
  const int r  = lane & 15;
  const int kh = lane >> 4;

  // This thread fills rows rbase+32i at column segment seg (8 bf16 = 16B).
  const int seg   = threadIdx.x & 7;
  const int rbase = threadIdx.x >> 3;

  const unsigned short* aSrc[4];
  const unsigned short* bSrc[4];
#pragma unroll
  for (int i = 0; i < 4; ++i) {
    int row = rbase + 32 * i;
    int tok = list[min(tileM + row, cntE - 1)];
    aSrc[i] = xb + (size_t)tok * D_ + seg * 8;
    bSrc[i] = w1t + ((size_t)e * H_ + n0 + row) * D_ + seg * 8;
  }

  v8f acc[4][2];
#pragma unroll
  for (int a = 0; a < 4; ++a)
#pragma unroll
    for (int b = 0; b < 2; ++b) acc[a][b] = (v8f)(0.0f);

  auto issue = [&](int p, int kc) {
    unsigned short* Abuf = smem + (p ? 2 * TILE_ELEMS : 0);
    unsigned short* Bbuf = Abuf + TILE_ELEMS;
#pragma unroll
    for (int i = 0; i < 4; ++i)
      copy16_g2l(aSrc[i] + kc, &Abuf[(rbase + 32 * i) * LDK + seg * 8]);
#pragma unroll
    for (int i = 0; i < 4; ++i)
      copy16_g2l(bSrc[i] + kc, &Bbuf[(rbase + 32 * i) * LDK + seg * 8]);
  };

  issue(0, 0);
  int p = 0;
  for (int kc = 0; kc < D_; kc += BK, p ^= 1) {
    bool more = (kc + BK) < D_;
    if (more) { issue(p ^ 1, kc + BK); wait_async_le8(); }
    else      { wait_async_0(); }
    __syncthreads();

    const unsigned short* Ac = smem + (p ? 2 * TILE_ELEMS : 0);
    const unsigned short* Bc = Ac + TILE_ELEMS;
#pragma unroll
    for (int ks = 0; ks < BK; ks += 32) {
      Frag af[4], bfr[2];
#pragma unroll
      for (int a = 0; a < 4; ++a) {
        const unsigned short* q = &Ac[(wm * 64 + a * 16 + r) * LDK + ks + kh * 8];
        af[a].q[0] = *(const int4*)q;
        af[a].q[1] = *(const int4*)(q + 16);
      }
#pragma unroll
      for (int b = 0; b < 2; ++b) {
        const unsigned short* q = &Bc[(wn * 32 + b * 16 + r) * LDK + ks + kh * 8];
        bfr[b].q[0] = *(const int4*)q;
        bfr[b].q[1] = *(const int4*)(q + 16);
      }
#pragma unroll
      for (int a = 0; a < 4; ++a)
#pragma unroll
        for (int b = 0; b < 2; ++b)
          acc[a][b] = __builtin_amdgcn_wmma_f32_16x16x32_bf16(
              false, af[a].v, false, bfr[b].v, (short)0, acc[a][b], false, false);
    }
    __syncthreads();
  }

  // Epilogue: +b1, exact GELU, bf16 store into compacted Hbuf
#pragma unroll
  for (int a = 0; a < 4; ++a) {
#pragma unroll
    for (int b = 0; b < 2; ++b) {
      int ncol = n0 + wn * 32 + b * 16 + r;
      float bias = b1[e * H_ + ncol];
#pragma unroll
      for (int j = 0; j < 8; ++j) {
        int mloc = wm * 64 + a * 16 + kh * 8 + j;
        int pos = tileM + mloc;
        if (pos < cntE) {
          float v = gelu_exact(acc[a][b][j] + bias);
          hbuf[(size_t)(base + pos) * H_ + ncol] = f2bf(v);
        }
      }
    }
  }
}

// ---------------- GEMM2: out += w * (Hbuf @ W2T + b2), atomic scatter ----------------

__global__ __launch_bounds__(256)
void moe_gemm2(const unsigned short* __restrict__ hbuf, const unsigned short* __restrict__ w2t,
               const float* __restrict__ b2,
               const int* __restrict__ cnt, const int* __restrict__ offs,
               const int* __restrict__ tokenId, const float* __restrict__ tokenW,
               float* __restrict__ out) {
  const int e = blockIdx.z;
  const int cntE = cnt[e];
  const int tileM = blockIdx.x * BM;
  if (tileM >= cntE) return;
  const int n0 = blockIdx.y * BN;
  const int base = offs[e];

  __shared__ __align__(16) unsigned short smem[4 * TILE_ELEMS];

  const int wave = threadIdx.x >> 5;
  const int lane = threadIdx.x & 31;
  const int wm = wave >> 2;
  const int wn = wave & 3;
  const int r  = lane & 15;
  const int kh = lane >> 4;

  const int seg   = threadIdx.x & 7;
  const int rbase = threadIdx.x >> 3;

  const unsigned short* aSrc[4];
  const unsigned short* bSrc[4];
#pragma unroll
  for (int i = 0; i < 4; ++i) {
    int row = rbase + 32 * i;
    int pos = min(tileM + row, cntE - 1);
    aSrc[i] = hbuf + (size_t)(base + pos) * H_ + seg * 8;
    bSrc[i] = w2t + ((size_t)e * D_ + n0 + row) * H_ + seg * 8;
  }

  v8f acc[4][2];
#pragma unroll
  for (int a = 0; a < 4; ++a)
#pragma unroll
    for (int b = 0; b < 2; ++b) acc[a][b] = (v8f)(0.0f);

  auto issue = [&](int p, int kc) {
    unsigned short* Abuf = smem + (p ? 2 * TILE_ELEMS : 0);
    unsigned short* Bbuf = Abuf + TILE_ELEMS;
#pragma unroll
    for (int i = 0; i < 4; ++i)
      copy16_g2l(aSrc[i] + kc, &Abuf[(rbase + 32 * i) * LDK + seg * 8]);
#pragma unroll
    for (int i = 0; i < 4; ++i)
      copy16_g2l(bSrc[i] + kc, &Bbuf[(rbase + 32 * i) * LDK + seg * 8]);
  };

  issue(0, 0);
  int p = 0;
  for (int kc = 0; kc < H_; kc += BK, p ^= 1) {
    bool more = (kc + BK) < H_;
    if (more) { issue(p ^ 1, kc + BK); wait_async_le8(); }
    else      { wait_async_0(); }
    __syncthreads();

    const unsigned short* Ac = smem + (p ? 2 * TILE_ELEMS : 0);
    const unsigned short* Bc = Ac + TILE_ELEMS;
#pragma unroll
    for (int ks = 0; ks < BK; ks += 32) {
      Frag af[4], bfr[2];
#pragma unroll
      for (int a = 0; a < 4; ++a) {
        const unsigned short* q = &Ac[(wm * 64 + a * 16 + r) * LDK + ks + kh * 8];
        af[a].q[0] = *(const int4*)q;
        af[a].q[1] = *(const int4*)(q + 16);
      }
#pragma unroll
      for (int b = 0; b < 2; ++b) {
        const unsigned short* q = &Bc[(wn * 32 + b * 16 + r) * LDK + ks + kh * 8];
        bfr[b].q[0] = *(const int4*)q;
        bfr[b].q[1] = *(const int4*)(q + 16);
      }
#pragma unroll
      for (int a = 0; a < 4; ++a)
#pragma unroll
        for (int b = 0; b < 2; ++b)
          acc[a][b] = __builtin_amdgcn_wmma_f32_16x16x32_bf16(
              false, af[a].v, false, bfr[b].v, (short)0, acc[a][b], false, false);
    }
    __syncthreads();
  }

#pragma unroll
  for (int a = 0; a < 4; ++a) {
#pragma unroll
    for (int b = 0; b < 2; ++b) {
      int ncol = n0 + wn * 32 + b * 16 + r;
      float bias = b2[e * D_ + ncol];
#pragma unroll
      for (int j = 0; j < 8; ++j) {
        int mloc = wm * 64 + a * 16 + kh * 8 + j;
        int pos = tileM + mloc;
        if (pos < cntE) {
          int   tok = tokenId[e * MAXT + pos];
          float w   = tokenW [e * MAXT + pos];
          unsafeAtomicAdd(&out[(size_t)tok * D_ + ncol], (acc[a][b][j] + bias) * w);
        }
      }
    }
  }
}

// ---------------- launcher ----------------

extern "C" void kernel_launch(void* const* d_in, const int* in_sizes, int n_in,
                              void* d_out, int out_size, void* d_ws, size_t ws_size,
                              hipStream_t stream) {
  const float* x  = (const float*)d_in[0];  // [B,S,D]
  const float* Wr = (const float*)d_in[1];  // [D,E]
  const float* br = (const float*)d_in[2];  // [E]
  const float* W1 = (const float*)d_in[3];  // [E,D,H]
  const float* b1 = (const float*)d_in[4];  // [E,H]
  const float* W2 = (const float*)d_in[5];  // [E,H,D]
  const float* b2 = (const float*)d_in[6];  // [E,D]
  float* out = (float*)d_out;

  char* ws = (char*)d_ws;
  int*   cnt     = (int*)ws;                         // E ints
  int*   offs    = (int*)(ws + 64);                  // E+1 ints
  int*   tokenId = (int*)(ws + 256);                 // E*MAXT
  float* tokenW  = (float*)(ws + 256 + (size_t)E_ * MAXT * 4);
  unsigned short* xb   = (unsigned short*)(ws + 256 + (size_t)E_ * MAXT * 8);
  unsigned short* w1t  = xb  + (size_t)NT * D_;
  unsigned short* w2t  = w1t + (size_t)E_ * H_ * D_;
  unsigned short* hbuf = w2t + (size_t)E_ * D_ * H_;

  hipLaunchKernelGGL(moe_zero_cnt, dim3(1), dim3(32), 0, stream, cnt);
  hipLaunchKernelGGL(moe_zero_f32, dim3(4096), dim3(256), 0, stream, out, (long long)out_size);
  hipLaunchKernelGGL(moe_cvt_x,  dim3(4096), dim3(256), 0, stream, x,  xb,  (long long)NT * D_);
  hipLaunchKernelGGL(moe_cvt_w1, dim3(4096), dim3(256), 0, stream, W1, w1t, (long long)E_ * H_ * D_);
  hipLaunchKernelGGL(moe_cvt_w2, dim3(4096), dim3(256), 0, stream, W2, w2t, (long long)E_ * D_ * H_);

  hipLaunchKernelGGL(moe_router, dim3(NT / 8), dim3(256), 0, stream,
                     x, Wr, br, cnt, tokenId, tokenW);
  hipLaunchKernelGGL(moe_scan, dim3(1), dim3(32), 0, stream, cnt, offs);

  hipLaunchKernelGGL(moe_gemm1, dim3(NT / BM, H_ / BN, E_), dim3(256), 0, stream,
                     xb, w1t, b1, cnt, offs, tokenId, hbuf);
  hipLaunchKernelGGL(moe_gemm2, dim3(NT / BM, D_ / BN, E_), dim3(256), 0, stream,
                     hbuf, w2t, b2, cnt, offs, tokenId, tokenW, out);
}
